// LSTM_EncoderDecoder_65609920414414
// MI455X (gfx1250) — matrix operand order repaired and checked by reading, not compiled, not run on hardware
//
#include <hip/hip_runtime.h>
#include <hip/hip_bf16.h>

// ---------------------------------------------------------------------------
// LSTM encoder-decoder for MI455X (gfx1250, wave32, WMMA).
//   I=64, H=512, S=256, B=512, T=128.
//
// One fused kernel per timestep: each wave computes the 16x16 (batch x hidden)
// patch of ALL FOUR gates (columns j, j+512, j+1024, j+1536 of the [512x2048]
// gates GEMM) with v_wmma_f32_16x16x32_f16 + fp32 accumulation, then applies
// sigmoid/tanh + cell update in registers, updating fp32 c and writing the new
// h directly in packed-f16 WMMA A-fragment layout for the next step.
//
// B operands (Whh/Wih) are packed gate-interleaved and j-tile-major so each
// workgroup streams ONE contiguous 72 KB region per step (L2-resident; the
// 8 waves of a block share it via WGP$). WGP-scope (scope 0) prefetches via
// inline asm run an 8 KB rolling window ahead of the loads so the dependent
// load->WMMA chain hits WGP$ instead of GL2. h ping-pongs between two packed
// buffers. Workspace budget: ~22.6 MB.
// ---------------------------------------------------------------------------

typedef __attribute__((ext_vector_type(16))) _Float16 v16h;
typedef __attribute__((ext_vector_type(8)))  float    v8f;

namespace {
constexpr int kI  = 64;
constexpr int kH  = 512;
constexpr int kS  = 256;
constexpr int kB  = 512;
constexpr int kT  = 128;
constexpr int kNG = 4 * kH;            // 2048 gate columns
constexpr int kKC = (kH + kI) / 32;    // 18 K-chunks (16 from h, 2 from x)
}

// ---- WMMA wrapper (codegen-confirmed builtin, 8 args) ----------------------
__device__ __forceinline__ v8f wmma_f16(v16h a, v16h b, v8f c) {
  return __builtin_amdgcn_wmma_f32_16x16x32_f16(
      /*neg_a=*/false, a, /*neg_b=*/false, b,
      /*c_mod=*/(short)0, c, /*reuse_a=*/false, /*reuse_b=*/false);
}

// ---- WGP-scope prefetch (scope 0 = pull into all cache levels on miss) -----
// __builtin_prefetch can only reach SE/DEV scope (GL2-only); inline asm with
// no scope modifier encodes scope 0 (WGP). Non-speculative: address must be
// valid — all call sites stay inside the packed workspace buffers.
__device__ __forceinline__ void prefetch_wgp(const void* p) {
  asm volatile("global_prefetch_b8 %0, off" :: "v"(p));
}

// ---- Fast activations (v_exp_f32 path) -------------------------------------
__device__ __forceinline__ float sigmoid_f(float x) {
  return 1.0f / (1.0f + __expf(-x));
}
__device__ __forceinline__ float tanh_f(float x) {
  return 2.0f / (1.0f + __expf(-2.0f * x)) - 1.0f;
}

// ---- Packed layouts (ISA 7.12.2) -------------------------------------------
// A-matrix 16x32 f16 fragment: lane l holds M=l&15, group g=l>>4. For (m,k):
__device__ __forceinline__ size_t a_pack_idx(int m, int k, int Mtot) {
  int kc = k >> 5, kk = k & 31;
  int g = kk >> 4;
  int v = (kk >> 1) & 7;
  return ((((size_t)kc * Mtot + m) * 2 + g) * 8 + v) * 2 + (kk & 1);
}
// Generic B-matrix 32x16 f16 fragment (used for lin_W only): lane l holds
// N=l&15, g=l>>4 covers K halves 0-15 / 16-31.
__device__ __forceinline__ size_t b_pack_idx(int n, int k, int Ntot) {
  int kc = k >> 5, kk = k & 31;
  int g = kk >> 4;
  int v = (kk >> 1) & 7;
  return ((((size_t)kc * Ntot + n) * 2 + g) * 8 + v) * 2 + (kk & 1);
}

// ---- Prologue: generic pack fp32 W[Ntot][K] -> B-fragment layout (lin_W) ---
__global__ __launch_bounds__(256) void pack_w_kernel(
    const float* __restrict__ W, _Float16* __restrict__ PB, int Ntot, int K) {
  int idx = blockIdx.x * blockDim.x + threadIdx.x;
  if (idx >= Ntot * K) return;
  int n = idx / K, k = idx - n * K;
  PB[b_pack_idx(n, k, Ntot)] = (_Float16)W[idx];
}

// ---- Prologue: gate-interleaved pack for Whh/Wih [2048][K] -----------------
// Layout: [jt=0..31][kc=0..K/32-1][gate=0..3][lane=0..31][16 halves]
// -> for a fixed j-tile the whole K sweep is contiguous (streamed per block).
__global__ __launch_bounds__(256) void pack_wg_kernel(
    const float* __restrict__ W, _Float16* __restrict__ PB, int K) {
  int idx = blockIdx.x * blockDim.x + threadIdx.x;
  if (idx >= kNG * K) return;
  int n = idx / K, k = idx - n * K;
  int gate = n >> 9, j = n & (kH - 1);
  int jt = j >> 4, nr = j & 15;
  int kc = k >> 5, kk = k & 31;
  int lg = kk >> 4, v = (kk >> 1) & 7, bit = kk & 1;
  int lane = lg * 16 + nr;
  size_t o = ((((size_t)jt * (K / 32) + kc) * 4 + gate) * 32 + lane) * 16
             + v * 2 + bit;
  PB[o] = (_Float16)W[idx];
}

// ---- Prologue: pack all encoder inputs [S][B][I] -> A-fragment layout ------
__global__ __launch_bounds__(256) void pack_x_kernel(
    const float* __restrict__ X, _Float16* __restrict__ PA, int total) {
  int idx = blockIdx.x * blockDim.x + threadIdx.x;
  if (idx >= total) return;
  int t = idx / (kB * kI);
  int r = idx - t * (kB * kI);
  int m = r / kI, k = r - m * kI;
  PA[(size_t)t * (kB * kI) + a_pack_idx(m, k, kB)] = (_Float16)X[idx];
}

// ---- Prologue: h = 0 (packed f16), c = 0 (fp32) ----------------------------
__global__ __launch_bounds__(256) void init_state_kernel(
    float* __restrict__ c, _Float16* __restrict__ PAh) {
  int idx = blockIdx.x * blockDim.x + threadIdx.x;   // 0 .. B*H-1
  c[idx] = 0.0f;
  PAh[idx] = (_Float16)0.0f;
}

// ---- Fused LSTM step --------------------------------------------------------
// 1024 waves: 32 M-strips (16 batch rows) x 32 j-tiles (16 hidden units).
// Wave accumulators: gate columns {j, j+512, j+1024, j+1536} -> i, f, g, o.
__global__ __launch_bounds__(256) void lstm_step_kernel(
    const _Float16* __restrict__ PAh,   // packed h_in  [K=512, M=512]
    const _Float16* __restrict__ PAx,   // packed x     [K=64,  M=512]
    const _Float16* __restrict__ PBh,   // gate-packed Whh^T (jt-major)
    const _Float16* __restrict__ PBx,   // gate-packed Wih^T (jt-major)
    const float* __restrict__ bih, const float* __restrict__ bhh,
    float* __restrict__ c,              // [512][512] fp32 (in/out)
    _Float16* __restrict__ PAh_out)     // packed h_out (next step's A)
{
  const int lane = threadIdx.x & 31;
  const int wave = blockIdx.x * 8 + (threadIdx.x >> 5);
  const int m0 = (wave & 31) << 4;
  const int j0 = (wave >> 5) << 4;
  const int lr = lane & 15;
  const int lg = lane >> 4;

  // A: per-lane 32B v16h slices, chunk stride 16384 halves.
  const size_t aoff = ((size_t)(m0 + lr) * 2 + lg) << 4;
  // B: j-tile-major contiguous stream; chunk stride 2048 halves (4 KB),
  // gate stride 512 halves (1 KB).
  const _Float16* bb_h = PBh + (size_t)j0 * 2048 + (size_t)lane * 16;
  const _Float16* bb_x = PBx + (size_t)j0 * 256  + (size_t)lane * 16;

  auto loadA = [&](int kc) -> v16h {
    const _Float16* base = (kc < kH / 32)
        ? (PAh + (size_t)kc * (kB * 32))
        : (PAx + (size_t)(kc - kH / 32) * (kB * 32));
    return *(const v16h*)(base + aoff);
  };
  auto loadB = [&](int kc, int gate) -> v16h {
    const _Float16* p = (kc < kH / 32)
        ? (bb_h + (size_t)kc * 2048 + (size_t)gate * 512)
        : (bb_x + (size_t)(kc - kH / 32) * 2048 + (size_t)gate * 512);
    return *(const v16h*)p;
  };

  v8f ai = {}, af = {}, ag = {}, ao = {};

  // Software pipeline: fragments for chunk kc+1 load during chunk kc's WMMAs.
  v16h aC = loadA(0);
  v16h b0 = loadB(0, 0), b1 = loadB(0, 1), b2 = loadB(0, 2), b3 = loadB(0, 3);

#pragma unroll
  for (int kc = 0; kc < kKC; ++kc) {
    v16h aN = aC, n0 = b0, n1 = b1, n2 = b2, n3 = b3;
    if (kc + 1 < kKC) {
      aN = loadA(kc + 1);
      n0 = loadB(kc + 1, 0);
      n1 = loadB(kc + 1, 1);
      n2 = loadB(kc + 1, 2);
      n3 = loadB(kc + 1, 3);
    }
    // WGP-scope prefetch running an 8 KB window ahead on the B stream and one
    // chunk ahead on the A stream: dependent load->WMMA chain hits WGP$.
    if (kc + 2 < kH / 32) {
      prefetch_wgp(bb_h + (size_t)(kc + 2) * 2048);
    }
    if (kc + 2 < kH / 32 && (kc & 1) == 0) {
      prefetch_wgp(PAh + (size_t)(kc + 2) * (kB * 32) + aoff);
    }
    ai = wmma_f16(aC, b0, ai);
    af = wmma_f16(aC, b1, af);
    ag = wmma_f16(aC, b2, ag);
    ao = wmma_f16(aC, b3, ao);
    aC = aN; b0 = n0; b1 = n1; b2 = n2; b3 = n3;
  }

  // ---- fused LSTM cell update (fp32 math) ----
  const int j = j0 + lr;
  const float bi_ = bih[j]            + bhh[j];
  const float bf_ = bih[j + kH]       + bhh[j + kH];
  const float bg_ = bih[j + 2 * kH]   + bhh[j + 2 * kH];
  const float bo_ = bih[j + 3 * kH]   + bhh[j + 3 * kH];

#pragma unroll
  for (int r = 0; r < 8; ++r) {
    const int m = m0 + r + (lg << 3);   // C/D layout: VGPR r -> row m0+r+8*lg
    const float i_ = sigmoid_f(ai[r] + bi_);
    const float f_ = sigmoid_f(af[r] + bf_);
    const float g_ = tanh_f(ag[r] + bg_);
    const float o_ = sigmoid_f(ao[r] + bo_);
    const size_t ci = (size_t)m * kH + j;
    const float cn = f_ * c[ci] + i_ * g_;
    c[ci] = cn;
    PAh_out[a_pack_idx(m, j, kB)] = (_Float16)(o_ * tanh_f(cn));
  }
}

// ---- Decoder projection: out = h@lin_W^T + lin_b; also feeds x back --------
// 128 waves: 32 M-strips x 4 N-tiles of 16.
__global__ __launch_bounds__(256) void proj_out_kernel(
    const _Float16* __restrict__ PAh,   // packed h    [K=512, M=512]
    const _Float16* __restrict__ PBl,   // packed linW [K=512, N=64]
    const float* __restrict__ lin_b,
    float* __restrict__ out_t,          // [512][64]
    _Float16* __restrict__ PAx_next)    // packed next x [K=64, M=512]
{
  const int lane = threadIdx.x & 31;
  const int wave = blockIdx.x * 8 + (threadIdx.x >> 5);
  const int m0 = (wave >> 2) << 4;
  const int n0 = (wave & 3) << 4;
  const int lr = lane & 15;
  const int lg = lane >> 4;

  const size_t aoff = ((size_t)(m0 + lr) * 2 + lg) << 4;
  const size_t boff = ((size_t)(n0 + lr) * 2 + lg) << 4;

  v8f acc = {};
  v16h aC = *(const v16h*)(PAh + aoff);
  v16h bC = *(const v16h*)(PBl + boff);
#pragma unroll
  for (int kc = 0; kc < kH / 32; ++kc) {
    v16h aN = aC, bN = bC;
    if (kc + 1 < kH / 32) {
      aN = *(const v16h*)(PAh + (size_t)(kc + 1) * (kB * 32) + aoff);
      bN = *(const v16h*)(PBl + (size_t)(kc + 1) * (kI * 32) + boff);
    }
    acc = wmma_f16(aC, bC, acc);
    aC = aN; bC = bN;
  }

  const float bias = lin_b[n0 + lr];
#pragma unroll
  for (int r = 0; r < 8; ++r) {
    const int m = m0 + r + (lg << 3);
    const float v = acc[r] + bias;
    out_t[(size_t)m * kI + n0 + lr] = v;
    PAx_next[a_pack_idx(m, n0 + lr, kB)] = (_Float16)v;
  }
}

// ---------------------------------------------------------------------------
extern "C" void kernel_launch(void* const* d_in, const int* in_sizes, int n_in,
                              void* d_out, int out_size, void* d_ws, size_t ws_size,
                              hipStream_t stream) {
  (void)in_sizes; (void)n_in; (void)out_size; (void)ws_size;

  const float* input_batch = (const float*)d_in[0];   // [S][B][I]
  const float* enc_Wih = (const float*)d_in[1];       // [4H][I]
  const float* enc_Whh = (const float*)d_in[2];       // [4H][H]
  const float* enc_bih = (const float*)d_in[3];
  const float* enc_bhh = (const float*)d_in[4];
  const float* dec_Wih = (const float*)d_in[5];
  const float* dec_Whh = (const float*)d_in[6];
  const float* dec_bih = (const float*)d_in[7];
  const float* dec_bhh = (const float*)d_in[8];
  const float* lin_W   = (const float*)d_in[9];       // [I][H]
  const float* lin_b   = (const float*)d_in[10];
  // d_in[11] = target_len (device scalar) -> fixed kT=128 at compile time.
  float* out = (float*)d_out;                         // [T][B][I]

  // ---- workspace carve-up (all region sizes are multiples of 256 B) ----
  char* w = (char*)d_ws;
  auto take = [&](size_t bytes) { char* p = w; w += bytes; return p; };
  _Float16* PB_eWhh = (_Float16*)take((size_t)kNG * kH * 2);     // 2 MB
  _Float16* PB_eWih = (_Float16*)take((size_t)kNG * kI * 2);     // 256 KB
  _Float16* PB_dWhh = (_Float16*)take((size_t)kNG * kH * 2);     // 2 MB
  _Float16* PB_dWih = (_Float16*)take((size_t)kNG * kI * 2);     // 256 KB
  _Float16* PB_lin  = (_Float16*)take((size_t)kI * kH * 2);      // 64 KB
  _Float16* PAx_all = (_Float16*)take((size_t)kS * kB * kI * 2); // 16 MB
  _Float16* PAx_dec = (_Float16*)take((size_t)kB * kI * 2);      // 64 KB
  _Float16* PA_hA   = (_Float16*)take((size_t)kB * kH * 2);      // 512 KB
  _Float16* PA_hB   = (_Float16*)take((size_t)kB * kH * 2);      // 512 KB
  float*    c_buf   = (float*)   take((size_t)kB * kH * 4);      // 1 MB

  // ---- prologue: pack weights + inputs, zero state ----
  pack_wg_kernel<<<(kNG * kH + 255) / 256, 256, 0, stream>>>(enc_Whh, PB_eWhh, kH);
  pack_wg_kernel<<<(kNG * kI + 255) / 256, 256, 0, stream>>>(enc_Wih, PB_eWih, kI);
  pack_wg_kernel<<<(kNG * kH + 255) / 256, 256, 0, stream>>>(dec_Whh, PB_dWhh, kH);
  pack_wg_kernel<<<(kNG * kI + 255) / 256, 256, 0, stream>>>(dec_Wih, PB_dWih, kI);
  pack_w_kernel<<<(kI * kH + 255) / 256, 256, 0, stream>>>(lin_W, PB_lin, kI, kH);
  pack_x_kernel<<<(kS * kB * kI + 255) / 256, 256, 0, stream>>>(
      input_batch, PAx_all, kS * kB * kI);
  init_state_kernel<<<(kB * kH) / 256, 256, 0, stream>>>(c_buf, PA_hA);

  const size_t xstride = (size_t)kB * kI;   // packed halves per timestep

  _Float16* hcur = PA_hA;   // h_t   (read)
  _Float16* hnxt = PA_hB;   // h_t+1 (write)

  // ---- encoder: 256 sequential fused steps ----
  for (int t = 0; t < kS; ++t) {
    lstm_step_kernel<<<128, 256, 0, stream>>>(
        hcur, PAx_all + (size_t)t * xstride, PB_eWhh, PB_eWih,
        enc_bih, enc_bhh, c_buf, hnxt);
    _Float16* tmp = hcur; hcur = hnxt; hnxt = tmp;
  }

  // ---- decoder: 128 autoregressive steps (x0 = input_batch[-1]) ----
  for (int t = 0; t < kT; ++t) {
    const _Float16* px = (t == 0) ? (PAx_all + (size_t)(kS - 1) * xstride)
                                  : PAx_dec;
    lstm_step_kernel<<<128, 256, 0, stream>>>(
        hcur, px, PB_dWhh, PB_dWih, dec_bih, dec_bhh, c_buf, hnxt);
    _Float16* tmp = hcur; hcur = hnxt; hnxt = tmp;
    proj_out_kernel<<<16, 256, 0, stream>>>(
        hcur, PB_lin, lin_b, out + (size_t)t * kB * kI, PAx_dec);
  }
}
